// GroupQueryAttention_82274393522289
// MI455X (gfx1250) — compile-verified, matrix operand order checked by reading, not run on hardware
//
#include <hip/hip_runtime.h>
#include <math.h>
#include <stdint.h>

typedef __bf16 v16bf __attribute__((ext_vector_type(16)));
typedef __bf16 bf16x8 __attribute__((ext_vector_type(8)));
typedef float  v8f    __attribute__((ext_vector_type(8)));
typedef int    v4i    __attribute__((ext_vector_type(4)));

#define T_DIM 2048
#define D_DIM 2048
#define H_NUM 16
#define G_NUM 4
#define HD_DIM 128

#define AS1 __attribute__((address_space(1)))
#define AS3 __attribute__((address_space(3)))

__device__ inline v8f zero8() {
    v8f z;
#pragma unroll
    for (int i = 0; i < 8; ++i) z[i] = 0.0f;
    return z;
}

// ---------------------------------------------------------------------------
// Async global->LDS staging helpers (CDNA5 GLOBAL_LOAD_ASYNC_TO_LDS_B128,
// ASYNCcnt). Parameter types per hipcc diagnostic: (v4i AS1*, v4i AS3*, ...).
// Falls back to a synchronous 16-byte copy if the builtin is absent.
// ---------------------------------------------------------------------------
#if defined(__gfx1250__) && __has_builtin(__builtin_amdgcn_global_load_async_to_lds_b128)
#define USE_ASYNC_LDS 1
#else
#define USE_ASYNC_LDS 0
#endif

__device__ inline void stage_b128(const __bf16* src, __bf16* dst_lds) {
#if USE_ASYNC_LDS
    __builtin_amdgcn_global_load_async_to_lds_b128(
        (AS1 v4i*)(uintptr_t)src,
        (AS3 v4i*)(uint32_t)(uintptr_t)dst_lds,
        0, 0);
#else
    *(bf16x8*)dst_lds = *(const bf16x8*)src;
#endif
}

__device__ inline void stage_wait() {
#if USE_ASYNC_LDS
#if __has_builtin(__builtin_amdgcn_s_wait_asynccnt)
    __builtin_amdgcn_s_wait_asynccnt(0);
#else
    asm volatile("s_wait_asynccnt 0" ::: "memory");
#endif
#endif
}

// ---------------------------------------------------------------------------
// f32 -> bf16 conversion
// ---------------------------------------------------------------------------
__global__ void f32_to_bf16_kernel(const float* __restrict__ in,
                                   __bf16* __restrict__ out, size_t n) {
    size_t i = (size_t)blockIdx.x * blockDim.x + threadIdx.x;
    if (i < n) out[i] = (__bf16)in[i];
}

// ---------------------------------------------------------------------------
// WMMA GEMM: C[M][N] = A[M][K] (bf16) * W[N][K]^T (bf16), f32 accumulate.
// Block (8 waves) computes a 128x128 macro-tile. B panel (128x32 bf16 = 8KB)
// is double-buffered in LDS via async copies; each wave owns a 16-row strip
// and 8 accumulators (8 WMMAs per K=32 step per single A-fragment load).
// ---------------------------------------------------------------------------
__global__ void gemm_bf16_wmma_lds(const __bf16* __restrict__ A,
                                   const __bf16* __restrict__ W,
                                   float* __restrict__ C,
                                   int M, int N, int K) {
    __shared__ __align__(16) __bf16 bTile[2][128 * 32];   // 2 x 8KB

    const int tid  = threadIdx.x;
    const int lane = tid & 31;
    const int wib  = tid >> 5;
    const int l15  = lane & 15;
    const int half = lane >> 4;

    const int nBlocks = N >> 7;
    const int mBlock  = blockIdx.x / nBlocks;
    const int nBlock  = blockIdx.x % nBlocks;
    const int m0 = (mBlock << 7) + (wib << 4);   // this wave's 16-row strip
    const int n0 = nBlock << 7;

    // Cooperative B staging: 512 b128 chunks per K-step, 2 per thread.
    const int srow = tid >> 2;            // 0..63   (and srow+64)
    const int ske  = (tid & 3) << 3;      // element offset 0,8,16,24
    const __bf16* gB = W + (size_t)n0 * K;

    // Prologue: stage K-step 0 into buffer 0
    stage_b128(gB + (size_t)srow * K + ske,        &bTile[0][srow * 32 + ske]);
    stage_b128(gB + (size_t)(srow + 64) * K + ske, &bTile[0][(srow + 64) * 32 + ske]);
    stage_wait();
    __syncthreads();

    const __bf16* aPtr = A + (size_t)(m0 + l15) * K + half * 8;

    v8f acc[8];
#pragma unroll
    for (int j = 0; j < 8; ++j) acc[j] = zero8();

    int buf = 0;
    for (int k0 = 0; k0 < K; k0 += 32) {
        // Prefetch next B panel into the other buffer (overlaps the WMMAs)
        if (k0 + 32 < K) {
            const __bf16* gBk = gB + k0 + 32;
            stage_b128(gBk + (size_t)srow * K + ske,
                       &bTile[buf ^ 1][srow * 32 + ske]);
            stage_b128(gBk + (size_t)(srow + 64) * K + ske,
                       &bTile[buf ^ 1][(srow + 64) * 32 + ske]);
        }

        __builtin_prefetch(aPtr + k0 + 128, 0, 3);
        bf16x8 alo = *(const bf16x8*)(aPtr + k0);
        bf16x8 ahi = *(const bf16x8*)(aPtr + k0 + 16);
        v16bf a;
#pragma unroll
        for (int i = 0; i < 8; ++i) { a[i] = alo[i]; a[i + 8] = ahi[i]; }

        const __bf16* bbase = &bTile[buf][0];
#pragma unroll
        for (int j = 0; j < 8; ++j) {
            bf16x8 blo = *(const bf16x8*)(bbase + (j * 16 + l15) * 32 + half * 8);
            bf16x8 bhi = *(const bf16x8*)(bbase + (j * 16 + l15) * 32 + 16 + half * 8);
            v16bf b;
#pragma unroll
            for (int i = 0; i < 8; ++i) { b[i] = blo[i]; b[i + 8] = bhi[i]; }
            acc[j] = __builtin_amdgcn_wmma_f32_16x16x32_bf16(
                false, a, false, b, (short)0, acc[j], false, false);
        }

        stage_wait();      // drain this iteration's async stores to LDS
        __syncthreads();   // make buf^1 visible / protect buf re-use
        buf ^= 1;
    }

#pragma unroll
    for (int j = 0; j < 8; ++j) {
#pragma unroll
        for (int r = 0; r < 8; ++r) {
            C[(size_t)(m0 + r + 8 * half) * N + n0 + j * 16 + l15] = acc[j][r];
        }
    }
}

// ---------------------------------------------------------------------------
// Per-token RMSNorm + RoPE + scale + bf16 pack (+ V transpose).
// One block (256 threads) per token t.
// ---------------------------------------------------------------------------
__global__ void norm_rope_pack(const float* __restrict__ qf,   // [T][2048]
                               const float* __restrict__ kf,   // [T][512]
                               const float* __restrict__ vf,   // [T][512]
                               const float* __restrict__ cosT, // [T][128]
                               const float* __restrict__ sinT, // [T][128]
                               const float* __restrict__ qw,   // [2048]
                               const float* __restrict__ kw,   // [512]
                               __bf16* __restrict__ q_bf,      // [T][2048]
                               __bf16* __restrict__ k_bf,      // [T][512]
                               __bf16* __restrict__ vT_bf) {   // [G*128][T]
    const int t   = blockIdx.x;
    const int tid = threadIdx.x;
    __shared__ float red[256];

    const float* qrow = qf + (size_t)t * (H_NUM * HD_DIM);
    const float* krow = kf + (size_t)t * (G_NUM * HD_DIM);
    const float* vrow = vf + (size_t)t * (G_NUM * HD_DIM);

    float s = 0.f;
    for (int i = tid; i < H_NUM * HD_DIM; i += 256) { float v = qrow[i]; s += v * v; }
    red[tid] = s; __syncthreads();
    for (int off = 128; off > 0; off >>= 1) {
        if (tid < off) red[tid] += red[tid + off];
        __syncthreads();
    }
    const float qscale = rsqrtf(red[0] / (float)(H_NUM * HD_DIM) + 1e-6f);
    __syncthreads();

    s = 0.f;
    for (int i = tid; i < G_NUM * HD_DIM; i += 256) { float v = krow[i]; s += v * v; }
    red[tid] = s; __syncthreads();
    for (int off = 128; off > 0; off >>= 1) {
        if (tid < off) red[tid] += red[tid + off];
        __syncthreads();
    }
    const float kscale = rsqrtf(red[0] / (float)(G_NUM * HD_DIM) + 1e-6f);

    const float* c  = cosT + (size_t)t * HD_DIM;
    const float* sn = sinT + (size_t)t * HD_DIM;
    const float attnScale = 0.088388347648318447f; // 128^-0.5

    for (int i = tid; i < H_NUM * HD_DIM; i += 256) {
        int hd = i & (HD_DIM - 1);
        float xv = qrow[i] * qscale * qw[i];
        float pr = (hd < 64) ? qrow[i + 64] * qscale * qw[i + 64]
                             : qrow[i - 64] * qscale * qw[i - 64];
        float rot = (hd < 64) ? -pr : pr;
        q_bf[(size_t)t * (H_NUM * HD_DIM) + i] =
            (__bf16)((xv * c[hd] + rot * sn[hd]) * attnScale);
    }
    for (int i = tid; i < G_NUM * HD_DIM; i += 256) {
        int hd = i & (HD_DIM - 1);
        float xv = krow[i] * kscale * kw[i];
        float pr = (hd < 64) ? krow[i + 64] * kscale * kw[i + 64]
                             : krow[i - 64] * kscale * kw[i - 64];
        float rot = (hd < 64) ? -pr : pr;
        k_bf[(size_t)t * (G_NUM * HD_DIM) + i] = (__bf16)(xv * c[hd] + rot * sn[hd]);
    }
    for (int i = tid; i < G_NUM * HD_DIM; i += 256) {
        vT_bf[(size_t)i * T_DIM + t] = (__bf16)vrow[i];
    }
}

// ---------------------------------------------------------------------------
// Flash attention, one wave per (head, 16-query tile). Causal, online softmax.
// blockDim = 256 (8 waves). Key steps of 32.
// ---------------------------------------------------------------------------
__global__ void flash_attn_wmma(const __bf16* __restrict__ q_bf,  // [T][2048]
                                const __bf16* __restrict__ k_bf,  // [T][512]
                                const __bf16* __restrict__ vT_bf, // [G*128][T]
                                __bf16* __restrict__ ctx_bf) {    // [T][2048]
    __shared__ __align__(16) __bf16 lds_p[8][16 * 32];

    const int wib  = threadIdx.x >> 5;
    const int lane = threadIdx.x & 31;
    const int task = blockIdx.x * 8 + wib;         // H * (T/16) tasks
    const int qTile = task & ((T_DIM >> 4) - 1);
    const int head  = task >> 7;                    // task / (T/16)
    const int g     = head >> 2;                    // GS = 4
    const int half  = lane >> 4;
    const int l15   = lane & 15;
    const int q0    = qTile << 4;

    __bf16* lds = &lds_p[wib][0];

    // Load 16x128 Q tile as 4 A-fragments (K=32 each)
    const __bf16* qPtr = q_bf + (size_t)(q0 + l15) * (H_NUM * HD_DIM)
                              + head * HD_DIM + half * 8;
    v16bf qfrag[4];
#pragma unroll
    for (int cI = 0; cI < 4; ++cI) {
        bf16x8 lo = *(const bf16x8*)(qPtr + cI * 32);
        bf16x8 hi = *(const bf16x8*)(qPtr + cI * 32 + 16);
#pragma unroll
        for (int i = 0; i < 8; ++i) { qfrag[cI][i] = lo[i]; qfrag[cI][i + 8] = hi[i]; }
    }

    float mrow[8], lrow[8];
    v8f acc[8];
#pragma unroll
    for (int r = 0; r < 8; ++r) { mrow[r] = -INFINITY; lrow[r] = 0.f; acc[r] = zero8(); }

    const int kEnd = q0 + 16;
    for (int k0 = 0; k0 < kEnd; k0 += 32) {
        // ---- S = Q @ K^T for 16x32 keys (two 16x16 D tiles) ----
        v8f s0 = zero8(), s1 = zero8();
        const __bf16* kPtr0 = k_bf + (size_t)(k0 + l15) * (G_NUM * HD_DIM)
                                   + g * HD_DIM + half * 8;
        const __bf16* kPtr1 = kPtr0 + (size_t)16 * (G_NUM * HD_DIM);
#pragma unroll
        for (int cI = 0; cI < 4; ++cI) {
            bf16x8 b0lo = *(const bf16x8*)(kPtr0 + cI * 32);
            bf16x8 b0hi = *(const bf16x8*)(kPtr0 + cI * 32 + 16);
            bf16x8 b1lo = *(const bf16x8*)(kPtr1 + cI * 32);
            bf16x8 b1hi = *(const bf16x8*)(kPtr1 + cI * 32 + 16);
            v16bf b0, b1;
#pragma unroll
            for (int i = 0; i < 8; ++i) {
                b0[i] = b0lo[i]; b0[i + 8] = b0hi[i];
                b1[i] = b1lo[i]; b1[i + 8] = b1hi[i];
            }
            s0 = __builtin_amdgcn_wmma_f32_16x16x32_bf16(
                false, qfrag[cI], false, b0, (short)0, s0, false, false);
            s1 = __builtin_amdgcn_wmma_f32_16x16x32_bf16(
                false, qfrag[cI], false, b1, (short)0, s1, false, false);
        }

        // ---- online softmax, per row (rows r+8*half across VGPRs) ----
        float rowscale[8];
#pragma unroll
        for (int r = 0; r < 8; ++r) {
            int qi   = q0 + r + 8 * half;
            int key0 = k0 + l15;
            int key1 = key0 + 16;
            float v0 = (key0 <= qi) ? s0[r] : -INFINITY;
            float v1 = (key1 <= qi) ? s1[r] : -INFINITY;
            float mx = fmaxf(v0, v1);
#pragma unroll
            for (int off = 8; off > 0; off >>= 1)
                mx = fmaxf(mx, __shfl_xor(mx, off, 32));
            float mnew = fmaxf(mrow[r], mx);
            float corr = __expf(mrow[r] - mnew);
            float p0 = __expf(v0 - mnew);
            float p1 = __expf(v1 - mnew);
            float ps = p0 + p1;
#pragma unroll
            for (int off = 8; off > 0; off >>= 1)
                ps += __shfl_xor(ps, off, 32);
            mrow[r] = mnew;
            lrow[r] = lrow[r] * corr + ps;
            rowscale[r] = corr;
            int m = r + 8 * half;
            lds[m * 32 + l15]      = (__bf16)p0;
            lds[m * 32 + 16 + l15] = (__bf16)p1;
        }
#pragma unroll
        for (int cI = 0; cI < 8; ++cI)
#pragma unroll
            for (int r = 0; r < 8; ++r) acc[cI][r] *= rowscale[r];

        // ---- read P back in A-fragment layout (16x32, K=32 keys) ----
        bf16x8 plo = *(const bf16x8*)(lds + l15 * 32 + half * 8);
        bf16x8 phi = *(const bf16x8*)(lds + l15 * 32 + 16 + half * 8);
        v16bf pfrag;
#pragma unroll
        for (int i = 0; i < 8; ++i) { pfrag[i] = plo[i]; pfrag[i + 8] = phi[i]; }

        // ---- ctx += P @ V : 8 column chunks of 16 over HD=128 ----
#pragma unroll
        for (int cI = 0; cI < 8; ++cI) {
            const __bf16* vp = vT_bf + (size_t)(g * HD_DIM + cI * 16 + l15) * T_DIM
                                     + k0 + half * 8;
            bf16x8 vlo = *(const bf16x8*)vp;
            bf16x8 vhi = *(const bf16x8*)(vp + 16);
            v16bf vfrag;
#pragma unroll
            for (int i = 0; i < 8; ++i) { vfrag[i] = vlo[i]; vfrag[i + 8] = vhi[i]; }
            acc[cI] = __builtin_amdgcn_wmma_f32_16x16x32_bf16(
                false, pfrag, false, vfrag, (short)0, acc[cI], false, false);
        }
    }

    // ---- normalize and write ctx (bf16, [T][H*HD]) ----
#pragma unroll
    for (int r = 0; r < 8; ++r) {
        float inv = 1.0f / lrow[r];
        int qi = q0 + r + 8 * half;
#pragma unroll
        for (int cI = 0; cI < 8; ++cI) {
            ctx_bf[(size_t)qi * (H_NUM * HD_DIM) + head * HD_DIM + cI * 16 + l15] =
                (__bf16)(acc[cI][r] * inv);
        }
    }
}

// ---------------------------------------------------------------------------
// Host launcher
// ---------------------------------------------------------------------------
extern "C" void kernel_launch(void* const* d_in, const int* in_sizes, int n_in,
                              void* d_out, int out_size, void* d_ws, size_t ws_size,
                              hipStream_t stream) {
    (void)in_sizes; (void)n_in; (void)out_size; (void)ws_size;
    const int T = T_DIM, D = D_DIM;
    const int NQ = H_NUM * HD_DIM;   // 2048
    const int NKV = G_NUM * HD_DIM;  // 512

    const float* x    = (const float*)d_in[0];
    // d_in[1] = mask (causal, reimplemented in-kernel)
    const float* cosT = (const float*)d_in[2];
    const float* sinT = (const float*)d_in[3];
    const float* Wq   = (const float*)d_in[4];
    const float* Wk   = (const float*)d_in[5];
    const float* Wv   = (const float*)d_in[6];
    const float* Wo   = (const float*)d_in[7];
    const float* qw   = (const float*)d_in[8];
    const float* kw   = (const float*)d_in[9];

    char* ws = (char*)d_ws;
    size_t off = 0;
    auto alloc = [&](size_t bytes) -> void* {
        void* p = ws + off;
        off += (bytes + 255) & ~(size_t)255;
        return p;
    };

    __bf16* x_bf   = (__bf16*)alloc((size_t)T * D * 2);
    __bf16* Wq_bf  = (__bf16*)alloc((size_t)NQ * D * 2);
    __bf16* Wk_bf  = (__bf16*)alloc((size_t)NKV * D * 2);
    __bf16* Wv_bf  = (__bf16*)alloc((size_t)NKV * D * 2);
    __bf16* Wo_bf  = (__bf16*)alloc((size_t)D * NQ * 2);
    float*  qf     = (float*)alloc((size_t)T * NQ * 4);
    float*  kf     = (float*)alloc((size_t)T * NKV * 4);
    float*  vf     = (float*)alloc((size_t)T * NKV * 4);
    __bf16* q_bf   = (__bf16*)alloc((size_t)T * NQ * 2);
    __bf16* k_bf   = (__bf16*)alloc((size_t)T * NKV * 2);
    __bf16* vT_bf  = (__bf16*)alloc((size_t)NKV * T * 2 + 256);
    __bf16* ctx_bf = (__bf16*)alloc((size_t)T * NQ * 2);

    auto cvt = [&](const float* src, __bf16* dst, size_t n) {
        int blocks = (int)((n + 255) / 256);
        f32_to_bf16_kernel<<<blocks, 256, 0, stream>>>(src, dst, n);
    };
    cvt(x,  x_bf,  (size_t)T * D);
    cvt(Wq, Wq_bf, (size_t)NQ * D);
    cvt(Wk, Wk_bf, (size_t)NKV * D);
    cvt(Wv, Wv_bf, (size_t)NKV * D);
    cvt(Wo, Wo_bf, (size_t)D * NQ);

    auto gemm = [&](const __bf16* A, const __bf16* W, float* C,
                    int M, int N, int K) {
        int blocks = (M >> 7) * (N >> 7);   // 128x128 macro-tiles
        gemm_bf16_wmma_lds<<<blocks, 256, 0, stream>>>(A, W, C, M, N, K);
    };
    gemm(x_bf, Wq_bf, qf, T, NQ, D);
    gemm(x_bf, Wk_bf, kf, T, NKV, D);
    gemm(x_bf, Wv_bf, vf, T, NKV, D);

    norm_rope_pack<<<T, 256, 0, stream>>>(qf, kf, vf, cosT, sinT, qw, kw,
                                          q_bf, k_bf, vT_bf);

    int attnTasks = H_NUM * (T >> 4);   // 2048 waves
    flash_attn_wmma<<<attnTasks / 8, 256, 0, stream>>>(q_bf, k_bf, vT_bf, ctx_bf);

    gemm(ctx_bf, Wo_bf, (float*)d_out, T, D, NQ);
}
